// GATClassifier_85564338471313
// MI455X (gfx1250) — matrix operand order repaired and checked by reading, not compiled, not run on hardware
//
#include <hip/hip_runtime.h>
#include <stdint.h>

#define NNODES  10000
#define NEDGES  160000
#define M_PAD   10240    // multiple of 128 (8 waves x 16 rows per block)

typedef __attribute__((ext_vector_type(16))) __bf16 v16bf;
typedef __attribute__((ext_vector_type(8)))  __bf16 v8bf;
typedef __attribute__((ext_vector_type(8)))  float  v8f;

// ---------------------------------------------------------------------------
// helpers
// ---------------------------------------------------------------------------
__device__ __forceinline__ unsigned short f2bf(float f) {
    unsigned int u = __float_as_uint(f);
    unsigned int r = u + 0x7FFFu + ((u >> 16) & 1u);   // round-to-nearest-even
    return (unsigned short)(r >> 16);
}

__device__ __forceinline__ float leaky(float x) { return x > 0.f ? x : 0.2f * x; }

// ---------------------------------------------------------------------------
// CSR build: degree count -> exclusive scan -> fill
// ---------------------------------------------------------------------------
__global__ void k_zero_i32(int* p, int n) {
    int i = blockIdx.x * blockDim.x + threadIdx.x;
    if (i < n) p[i] = 0;
}

__global__ void k_count_deg(const int* __restrict__ dst, int* __restrict__ deg, int e) {
    int i = blockIdx.x * blockDim.x + threadIdx.x;
    if (i < e) atomicAdd(&deg[dst[i]], 1);
}

__global__ void k_scan_offsets(const int* __restrict__ deg, int* __restrict__ offs, int n) {
    __shared__ int sh[1024];
    __shared__ int carry_s;
    int tid = threadIdx.x;
    if (tid == 0) carry_s = 0;
    __syncthreads();
    for (int base = 0; base < n; base += 1024) {
        int i = base + tid;
        int v = (i < n) ? deg[i] : 0;
        sh[tid] = v;
        __syncthreads();
        for (int off = 1; off < 1024; off <<= 1) {           // Hillis-Steele inclusive
            int t = (tid >= off) ? sh[tid - off] : 0;
            __syncthreads();
            sh[tid] += t;
            __syncthreads();
        }
        int carry = carry_s;
        if (i < n) offs[i] = carry + sh[tid] - v;            // exclusive
        __syncthreads();
        if (tid == 0) carry_s = carry + sh[1023];
        __syncthreads();
    }
    if (tid == 0) offs[n] = carry_s;
}

__global__ void k_copy_i32(const int* __restrict__ a, int* __restrict__ b, int n) {
    int i = blockIdx.x * blockDim.x + threadIdx.x;
    if (i < n) b[i] = a[i];
}

__global__ void k_fill_csr(const int* __restrict__ src, const int* __restrict__ dst,
                           int* __restrict__ cur, int* __restrict__ csr, int e) {
    int i = blockIdx.x * blockDim.x + threadIdx.x;
    if (i < e) {
        int pos = atomicAdd(&cur[dst[i]], 1);
        csr[pos] = src[i];
    }
}

// ---------------------------------------------------------------------------
// fp32 -> bf16 conversions
// ---------------------------------------------------------------------------
__global__ void k_convert_feat(const float* __restrict__ in, unsigned short* __restrict__ out,
                               long long count) {
    long long i = (long long)blockIdx.x * blockDim.x + threadIdx.x;
    if (i < count) out[i] = f2bf(in[i]);
}

// W [K, Nout] row-major -> WT bf16 [Nout, K] row-major
__global__ void k_convert_wT(const float* __restrict__ W, unsigned short* __restrict__ WT,
                             int K, int Nout) {
    long long total = (long long)K * Nout;
    long long i = (long long)blockIdx.x * blockDim.x + threadIdx.x;
    if (i >= total) return;
    int n = (int)(i / K);
    int k = (int)(i % K);
    WT[i] = f2bf(W[(long long)k * Nout + n]);
}

// ---------------------------------------------------------------------------
// WMMA GEMM: C[M_PAD, N] f32 = A[M_PAD, K] bf16 @ BT[N, K]^T bf16
// grid = (N/64, M_PAD/128), block = 256 (8 waves, each 16 rows x 64 cols)
// ---------------------------------------------------------------------------
__global__ __launch_bounds__(256)
void k_wmma_gemm(const __bf16* __restrict__ A, const __bf16* __restrict__ BT,
                 float* __restrict__ C, int N, int K) {
    int lane = threadIdx.x & 31;
    int wid  = threadIdx.x >> 5;
    int row0 = blockIdx.y * 128 + wid * 16;
    int col0 = blockIdx.x * 64;

    int ar  = lane & 15;            // A row within tile
    int kb  = (lane >> 4) << 4;     // A K sub-offset: 0 or 16
    int kbB = (lane >> 4) << 4;     // B K sub-offset: 0 or 16

    const __bf16* arow = A  + (long long)(row0 + ar) * K;
    const __bf16* bcol = BT + (long long)(col0 + (lane & 15)) * K + kbB;

    v8f c0 = {}, c1 = {}, c2 = {}, c3 = {};

    for (int k0 = 0; k0 < K; k0 += 32) {
        v8bf alo = *(const v8bf*)(arow + k0 + kb);
        v8bf ahi = *(const v8bf*)(arow + k0 + kb + 8);
        v16bf a;
#pragma unroll
        for (int i = 0; i < 8; ++i) { a[i] = alo[i]; a[i + 8] = ahi[i]; }

        v16bf b0 = *(const v16bf*)(bcol + k0);
        v16bf b1 = *(const v16bf*)(bcol + (long long)16 * K + k0);
        v16bf b2 = *(const v16bf*)(bcol + (long long)32 * K + k0);
        v16bf b3 = *(const v16bf*)(bcol + (long long)48 * K + k0);

        c0 = __builtin_amdgcn_wmma_f32_16x16x32_bf16(false, a, false, b0, (short)0, c0, false, false);
        c1 = __builtin_amdgcn_wmma_f32_16x16x32_bf16(false, a, false, b1, (short)0, c1, false, false);
        c2 = __builtin_amdgcn_wmma_f32_16x16x32_bf16(false, a, false, b2, (short)0, c2, false, false);
        c3 = __builtin_amdgcn_wmma_f32_16x16x32_bf16(false, a, false, b3, (short)0, c3, false, false);
    }

    // C/D layout: VGPR i, lanes 0-15 -> row i, lanes 16-31 -> row i+8; col = lane&15
    int orow = row0 + (lane >> 4) * 8;
    int ocol = col0 + (lane & 15);
    float* cp = C + (long long)orow * N + ocol;
#pragma unroll
    for (int i = 0; i < 8; ++i) {
        cp[0]  = c0[i];
        cp[16] = c1[i];
        cp[32] = c2[i];
        cp[48] = c3[i];
        cp += N;
    }
}

// ---------------------------------------------------------------------------
// per-(node,head) attention coefficients: alpha_src/dst[n,h] = <h[n,h,:], a_*[h,:]>
// one wave per (node, head)
// ---------------------------------------------------------------------------
template <int H, int C>
__global__ void k_alpha(const float* __restrict__ hlin, const float* __restrict__ a_src,
                        const float* __restrict__ a_dst, float* __restrict__ asrc,
                        float* __restrict__ adst, int n_nodes) {
    int wave = (blockIdx.x * blockDim.x + threadIdx.x) >> 5;
    int lane = threadIdx.x & 31;
    if (wave >= n_nodes * H) return;
    int n = wave / H, h = wave - n * H;
    const float* hp = hlin + (long long)n * H * C + h * C;
    const float* as = a_src + h * C;
    const float* ad = a_dst + h * C;
    float s = 0.f, d = 0.f;
    for (int c = lane; c < C; c += 32) {
        float v = hp[c];
        s += v * as[c];
        d += v * ad[c];
    }
#pragma unroll
    for (int off = 16; off > 0; off >>= 1) {
        s += __shfl_xor(s, off);
        d += __shfl_xor(d, off);
    }
    if (lane == 0) { asrc[wave] = s; adst[wave] = d; }
}

// ---------------------------------------------------------------------------
// CSR gather aggregation with segment softmax; bias + ReLU fused.
// one wave per (node, head); lane covers CPL channels (C = 32*CPL)
// ---------------------------------------------------------------------------
template <int H, int CPL>
__global__ void k_gat_aggregate(const float* __restrict__ hlin, const float* __restrict__ asrc,
                                const float* __restrict__ adst, const int* __restrict__ offs,
                                const int* __restrict__ csr, const float* __restrict__ bias,
                                float* __restrict__ out, int n_nodes) {
    const int C = CPL * 32;
    int wave = (blockIdx.x * blockDim.x + threadIdx.x) >> 5;
    int lane = threadIdx.x & 31;
    if (wave >= n_nodes * H) return;
    int n = wave / H, h = wave - n * H;
    int beg = offs[n], end = offs[n + 1], deg = end - beg;

    float acc[CPL];
#pragma unroll
    for (int p = 0; p < CPL; ++p) acc[p] = 0.f;

    if (deg > 0) {
        float ad = adst[n * H + h];
        // pass 1: segment max
        float m = -3.0e38f;
        for (int j = lane; j < deg; j += 32) {
            int s = csr[beg + j];
            m = fmaxf(m, leaky(asrc[s * H + h] + ad));
        }
#pragma unroll
        for (int off = 16; off > 0; off >>= 1) m = fmaxf(m, __shfl_xor(m, off));
        // pass 2: exp-sum
        float sum = 0.f;
        for (int j = lane; j < deg; j += 32) {
            int s = csr[beg + j];
            sum += __expf(leaky(asrc[s * H + h] + ad) - m);
        }
#pragma unroll
        for (int off = 16; off > 0; off >>= 1) sum += __shfl_xor(sum, off);
        float inv = 1.f / (sum + 1e-16f);
        // pass 3: weighted gather-accumulate (no atomics; lanes cover channels)
        for (int j = 0; j < deg; ++j) {
            int s = csr[beg + j];
            float w = __expf(leaky(asrc[s * H + h] + ad) - m) * inv;
            const float* hp = hlin + (long long)s * (H * C) + h * C + lane;
#pragma unroll
            for (int p = 0; p < CPL; ++p) acc[p] += w * hp[p * 32];
        }
    }
    float* op = out + (long long)n * (H * C) + h * C + lane;
    const float* bp = bias + h * C + lane;
#pragma unroll
    for (int p = 0; p < CPL; ++p) {
        float v = acc[p] + bp[p * 32];
        op[p * 32] = v > 0.f ? v : 0.f;
    }
}

// ---------------------------------------------------------------------------
// bias + ReLU (for linear layer), and final tiny classifier
// ---------------------------------------------------------------------------
__global__ void k_bias_relu(const float* __restrict__ in, const float* __restrict__ b,
                            float* __restrict__ out, int rows, int cols) {
    long long i = (long long)blockIdx.x * blockDim.x + threadIdx.x;
    if (i >= (long long)rows * cols) return;
    float v = in[i] + b[(int)(i % cols)];
    out[i] = v > 0.f ? v : 0.f;
}

__global__ void k_classifier(const float* __restrict__ h, const float* __restrict__ Wc,
                             const float* __restrict__ bc, float* __restrict__ out, int n) {
    int i = blockIdx.x * blockDim.x + threadIdx.x;
    if (i >= n * 3) return;
    int node = i / 3, o = i - node * 3;
    float s = bc[o];
    const float* hp = h + (long long)node * 512;
    for (int k = 0; k < 512; ++k) s += hp[k] * Wc[k * 3 + o];
    out[i] = s;
}

// ---------------------------------------------------------------------------
// launch
// ---------------------------------------------------------------------------
static inline int cdiv(long long a, long long b) { return (int)((a + b - 1) / b); }

extern "C" void kernel_launch(void* const* d_in, const int* in_sizes, int n_in,
                              void* d_out, int out_size, void* d_ws, size_t ws_size,
                              hipStream_t stream) {
    (void)in_sizes; (void)n_in; (void)out_size; (void)ws_size;

    const float* x     = (const float*)d_in[0];
    const int*   ei    = (const int*)d_in[1];      // [2, E]
    const float* W1    = (const float*)d_in[2];
    const float* as1   = (const float*)d_in[3];
    const float* ad1   = (const float*)d_in[4];
    const float* b1    = (const float*)d_in[5];
    const float* W2    = (const float*)d_in[6];
    const float* as2   = (const float*)d_in[7];
    const float* ad2   = (const float*)d_in[8];
    const float* b2    = (const float*)d_in[9];
    const float* W3    = (const float*)d_in[10];
    const float* as3   = (const float*)d_in[11];
    const float* ad3   = (const float*)d_in[12];
    const float* b3    = (const float*)d_in[13];
    const float* Wl    = (const float*)d_in[14];
    const float* bl    = (const float*)d_in[15];
    const float* Wc    = (const float*)d_in[16];
    const float* bc    = (const float*)d_in[17];
    float*       outp  = (float*)d_out;

    const int* e_src = ei;
    const int* e_dst = ei + NEDGES;

    // workspace layout
    char* ws = (char*)d_ws;
    size_t off = 0;
    auto walloc = [&](size_t bytes) -> void* {
        void* p = ws + off;
        off = (off + bytes + 255) & ~(size_t)255;
        return p;
    };
    int*            deg  = (int*)walloc((size_t)NNODES * 4);
    int*            offs = (int*)walloc((size_t)(NNODES + 1) * 4);
    int*            cur  = (int*)walloc((size_t)NNODES * 4);
    int*            csr  = (int*)walloc((size_t)NEDGES * 4);
    float*          asb  = (float*)walloc((size_t)NNODES * 8 * 4);
    float*          adb  = (float*)walloc((size_t)NNODES * 8 * 4);
    unsigned short* Abf  = (unsigned short*)walloc((size_t)M_PAD * 1024 * 2);
    unsigned short* WbfT = (unsigned short*)walloc((size_t)1024 * 1024 * 2);
    float*          bufA = (float*)walloc((size_t)M_PAD * 1024 * 4);
    float*          bufB = (float*)walloc((size_t)M_PAD * 1024 * 4);

    // ---- CSR build (deterministic, rebuilt every call) ----
    k_zero_i32<<<cdiv(NNODES, 256), 256, 0, stream>>>(deg, NNODES);
    k_count_deg<<<cdiv(NEDGES, 256), 256, 0, stream>>>(e_dst, deg, NEDGES);
    k_scan_offsets<<<1, 1024, 0, stream>>>(deg, offs, NNODES);
    k_copy_i32<<<cdiv(NNODES, 256), 256, 0, stream>>>(offs, cur, NNODES);
    k_fill_csr<<<cdiv(NEDGES, 256), 256, 0, stream>>>(e_src, e_dst, cur, csr, NEDGES);

    // ---- Layer 1: x[10000,128] @ W1[128,1024], H=8, C=128 ----
    k_convert_feat<<<cdiv((long long)NNODES * 128, 256), 256, 0, stream>>>(x, Abf, (long long)NNODES * 128);
    k_convert_wT<<<cdiv((long long)128 * 1024, 256), 256, 0, stream>>>(W1, WbfT, 128, 1024);
    {
        dim3 g(1024 / 64, M_PAD / 128);
        k_wmma_gemm<<<g, 256, 0, stream>>>((const __bf16*)Abf, (const __bf16*)WbfT, bufB, 1024, 128);
    }
    k_alpha<8, 128><<<cdiv((long long)NNODES * 8 * 32, 256), 256, 0, stream>>>(bufB, as1, ad1, asb, adb, NNODES);
    k_gat_aggregate<8, 4><<<cdiv((long long)NNODES * 8 * 32, 256), 256, 0, stream>>>(bufB, asb, adb, offs, csr, b1, bufA, NNODES);

    // ---- Layer 2: h[10000,1024] @ W2[1024,1024], H=8, C=128 ----
    k_convert_feat<<<cdiv((long long)NNODES * 1024, 256), 256, 0, stream>>>(bufA, Abf, (long long)NNODES * 1024);
    k_convert_wT<<<cdiv((long long)1024 * 1024, 256), 256, 0, stream>>>(W2, WbfT, 1024, 1024);
    {
        dim3 g(1024 / 64, M_PAD / 128);
        k_wmma_gemm<<<g, 256, 0, stream>>>((const __bf16*)Abf, (const __bf16*)WbfT, bufB, 1024, 1024);
    }
    k_alpha<8, 128><<<cdiv((long long)NNODES * 8 * 32, 256), 256, 0, stream>>>(bufB, as2, ad2, asb, adb, NNODES);
    k_gat_aggregate<8, 4><<<cdiv((long long)NNODES * 8 * 32, 256), 256, 0, stream>>>(bufB, asb, adb, offs, csr, b2, bufA, NNODES);

    // ---- Layer 3: h[10000,1024] @ W3[1024,512], H=1, C=512 ----
    k_convert_feat<<<cdiv((long long)NNODES * 1024, 256), 256, 0, stream>>>(bufA, Abf, (long long)NNODES * 1024);
    k_convert_wT<<<cdiv((long long)1024 * 512, 256), 256, 0, stream>>>(W3, WbfT, 1024, 512);
    {
        dim3 g(512 / 64, M_PAD / 128);
        k_wmma_gemm<<<g, 256, 0, stream>>>((const __bf16*)Abf, (const __bf16*)WbfT, bufB, 512, 1024);
    }
    k_alpha<1, 512><<<cdiv((long long)NNODES * 1 * 32, 256), 256, 0, stream>>>(bufB, as3, ad3, asb, adb, NNODES);
    k_gat_aggregate<1, 16><<<cdiv((long long)NNODES * 1 * 32, 256), 256, 0, stream>>>(bufB, asb, adb, offs, csr, b3, bufA, NNODES);

    // ---- Linear: h[10000,512] @ Wl[512,512] + bl, ReLU ----
    k_convert_feat<<<cdiv((long long)NNODES * 512, 256), 256, 0, stream>>>(bufA, Abf, (long long)NNODES * 512);
    k_convert_wT<<<cdiv((long long)512 * 512, 256), 256, 0, stream>>>(Wl, WbfT, 512, 512);
    {
        dim3 g(512 / 64, M_PAD / 128);
        k_wmma_gemm<<<g, 256, 0, stream>>>((const __bf16*)Abf, (const __bf16*)WbfT, bufB, 512, 512);
    }
    k_bias_relu<<<cdiv((long long)NNODES * 512, 256), 256, 0, stream>>>(bufB, bl, bufA, NNODES, 512);

    // ---- Classifier: h[10000,512] @ Wc[512,3] + bc ----
    k_classifier<<<cdiv((long long)NNODES * 3, 256), 256, 0, stream>>>(bufA, Wc, bc, outp, NNODES);
}
